// MultiHeadAttention_32487132627006
// MI455X (gfx1250) — matrix operand order repaired and checked by reading, not compiled, not run on hardware
//
#include <hip/hip_runtime.h>

// ---------------- problem constants ----------------
#define BQ   2
#define SQ   2048
#define DM   1024
#define NH   16
#define DH   64
#define MTOT (BQ * SQ)   // 4096 rows in model-dim GEMMs

typedef __bf16 bf16;
typedef __attribute__((ext_vector_type(16))) __bf16 v16bf;
typedef __attribute__((ext_vector_type(8)))  __bf16 v8bf;
typedef __attribute__((ext_vector_type(8)))  float  v8f;
typedef __attribute__((ext_vector_type(16))) float  v16f;

// concat two 8-wide bf16 vectors into a 16-wide WMMA fragment
__device__ __forceinline__ v16bf cat8(v8bf lo, v8bf hi) {
  return __builtin_shufflevector(lo, hi, 0,1,2,3,4,5,6,7,8,9,10,11,12,13,14,15);
}

__device__ __forceinline__ v8bf cvt8(v8f f) {
  v8bf r;
#pragma unroll
  for (int i = 0; i < 8; ++i) r[i] = (__bf16)f[i];
  return r;
}

__device__ __forceinline__ v16bf cvt16(v16f f) {
  v16bf r;
#pragma unroll
  for (int i = 0; i < 16; ++i) r[i] = (__bf16)f[i];
  return r;
}

// =====================================================================
// Kernel 1: fused projection GEMM  out = X @ W^T + b   (f32 in, bf16 out)
// One wave = one 16x16 output tile, K-loop in steps of 32 (bf16 WMMA K).
// scale folds the 1/8 attention scaling into Q; vtrans stores V as [B,H,Dh,S].
// =====================================================================
__global__ void proj_wmma(const float* __restrict__ X,   // [MTOT, DM]
                          const float* __restrict__ W,   // [DM, DM] (out,in)
                          const float* __restrict__ bias,// [DM]
                          bf16* __restrict__ out,
                          float scale, int vtrans) {
  const int lane = threadIdx.x & 31;
  const int wave = threadIdx.x >> 5;
  const int half = lane >> 4;
  const int lr   = lane & 15;

  const int tile = blockIdx.x * 8 + wave;
  const int tn   = tile & ((DM / 16) - 1);   // 64 tiles across N
  const int tm   = tile >> 6;
  const int m0 = tm * 16, n0 = tn * 16;

  const float* arow = X + (long)(m0 + lr) * DM;   // A: row = m0+lr
  const float* brow = W + (long)(n0 + lr) * DM;   // B: lane = output col

  v8f c;
#pragma unroll
  for (int r = 0; r < 8; ++r) c[r] = 0.0f;

  for (int kk = 0; kk < DM; kk += 32) {
    const int ka = kk + 16 * half;
    v16bf a = cat8(cvt8(*(const v8f*)(arow + ka)),
                   cvt8(*(const v8f*)(arow + ka + 8)));
    v16bf b = cvt16(*(const v16f*)(brow + kk + 16 * half));
    c = __builtin_amdgcn_wmma_f32_16x16x32_bf16(false, a, false, b,
                                                (short)0, c, false, false);
  }

  const float bv = bias[n0 + lr];
#pragma unroll
  for (int r = 0; r < 8; ++r) {
    const float val = (c[r] + bv) * scale;
    const int row = m0 + r + 8 * half;          // 0..4095  (b*S + s)
    const int col = n0 + lr;                    // 0..1023  (h*Dh + d)
    const int bb = row >> 11, ss = row & (SQ - 1);
    const int hh = col >> 6,  dd = col & (DH - 1);
    const long idx = vtrans
        ? ((long)((bb * NH + hh) * DH + dd) * SQ + ss)    // [B,H,Dh,S]
        : ((long)((bb * NH + hh) * SQ + ss) * DH + dd);   // [B,H,S,Dh]
    out[idx] = (__bf16)val;
  }
}

// =====================================================================
// Kernel 2: streaming (flash) attention, one wave = 16 query rows.
// Per 32-key step: 4 WMMAs for scores, online softmax in registers,
// P tile relayout via wave-private LDS, 4 WMMAs for P@V.
// =====================================================================
__global__ void flash_wmma(const bf16* __restrict__ Q,    // [B,H,S,Dh], pre-scaled by 1/8
                           const bf16* __restrict__ K,    // [B,H,S,Dh]
                           const bf16* __restrict__ Vt,   // [B,H,Dh,S]
                           const float* __restrict__ mask,// [S,S] broadcast
                           bf16* __restrict__ ctx) {      // [B*S, DM] bf16
  __shared__ __bf16 lds_p[8][16][32];   // per-wave 16x32 P tile (8 KB)

  const int lane = threadIdx.x & 31;
  const int wave = threadIdx.x >> 5;
  const int half = lane >> 4;
  const int lr   = lane & 15;

  const int nqb  = SQ / 128;                 // q-blocks per (b,h)
  const int bh   = blockIdx.x / nqb;
  const int qb   = blockIdx.x % nqb;
  const int qbase = qb * 128 + wave * 16;
  const int b = bh / NH, h = bh % NH;

  const bf16* Qp = Q  + (long)bh * SQ * DH;
  const bf16* Kp = K  + (long)bh * SQ * DH;
  const bf16* Vp = Vt + (long)bh * DH * SQ;

  // Q A-fragments for head-dim slices [0,32) and [32,64): loaded once
  const bf16* qrow = Qp + (long)(qbase + lr) * DH;
  const v16bf aq0 = cat8(*(const v8bf*)(qrow +      16 * half),
                         *(const v8bf*)(qrow + 8 +  16 * half));
  const v16bf aq1 = cat8(*(const v8bf*)(qrow + 32 + 16 * half),
                         *(const v8bf*)(qrow + 40 + 16 * half));

  float m[8], l[8];
  v8f acc[4];
#pragma unroll
  for (int r = 0; r < 8; ++r) { m[r] = -3.0e38f; l[r] = 0.0f; }
#pragma unroll
  for (int n = 0; n < 4; ++n)
#pragma unroll
    for (int r = 0; r < 8; ++r) acc[n][r] = 0.0f;

  for (int j = 0; j < SQ; j += 32) {
    if (j + 32 < SQ)  // pull next K tile toward L2/L0
      __builtin_prefetch(Kp + (long)(j + 32 + lr) * DH, 0, 1);

    // ---- scores: two 16x16 tiles (keys j..j+15, j+16..j+31) ----
    v8f s[2];
#pragma unroll
    for (int t = 0; t < 2; ++t) {
      const bf16* krow = Kp + (long)(j + 16 * t + lr) * DH;  // lane = key col
      const v16bf bk0 = *(const v16bf*)(krow +      16 * half);
      const v16bf bk1 = *(const v16bf*)(krow + 32 + 16 * half);
      v8f sz;
#pragma unroll
      for (int r = 0; r < 8; ++r) sz[r] = 0.0f;
      sz = __builtin_amdgcn_wmma_f32_16x16x32_bf16(false, aq0, false, bk0,
                                                   (short)0, sz, false, false);
      sz = __builtin_amdgcn_wmma_f32_16x16x32_bf16(false, aq1, false, bk1,
                                                   (short)0, sz, false, false);
      s[t] = sz;
    }

    // ---- mask add + online softmax (rows live across 16-lane halves) ----
#pragma unroll
    for (int r = 0; r < 8; ++r) {
      const long qg = qbase + r + 8 * half;
      s[0][r] += mask[qg * SQ + j + lr];
      s[1][r] += mask[qg * SQ + j + 16 + lr];
    }
#pragma unroll
    for (int r = 0; r < 8; ++r) {
      float tm = fmaxf(s[0][r], s[1][r]);
      tm = fmaxf(tm, __shfl_xor(tm, 1, 32));
      tm = fmaxf(tm, __shfl_xor(tm, 2, 32));
      tm = fmaxf(tm, __shfl_xor(tm, 4, 32));
      tm = fmaxf(tm, __shfl_xor(tm, 8, 32));
      const float mnew = fmaxf(m[r], tm);
      const float sc = __expf(m[r] - mnew);
      const float p0 = __expf(s[0][r] - mnew);
      const float p1 = __expf(s[1][r] - mnew);
      float rs = p0 + p1;
      rs += __shfl_xor(rs, 1, 32);
      rs += __shfl_xor(rs, 2, 32);
      rs += __shfl_xor(rs, 4, 32);
      rs += __shfl_xor(rs, 8, 32);
      l[r] = l[r] * sc + rs;
      m[r] = mnew;
#pragma unroll
      for (int n = 0; n < 4; ++n) acc[n][r] *= sc;
      // stash P in C-layout (row = r+8*half, col = lane + 16*t)
      lds_p[wave][r + 8 * half][lr]      = (__bf16)p0;
      lds_p[wave][r + 8 * half][lr + 16] = (__bf16)p1;
    }

    asm volatile("s_wait_dscnt 0x0" ::: "memory");  // wave-private LDS RAW fence

    // ---- reload P as A-fragment (16x32, contiguous K layout: k = 16*half + e) ----
    v16bf ap;
#pragma unroll
    for (int e = 0; e < 16; ++e) {
      const int col = 16 * half + e;
      ap[e] = lds_p[wave][lr][col];
    }

    // ---- ctx += P @ V  (V stored transposed: contiguous key runs) ----
#pragma unroll
    for (int n = 0; n < 4; ++n) {
      const bf16* vrow = Vp + (long)(n * 16 + lr) * SQ + j + 16 * half;
      const v16bf bv = *(const v16bf*)vrow;
      acc[n] = __builtin_amdgcn_wmma_f32_16x16x32_bf16(false, ap, false, bv,
                                                       (short)0, acc[n], false, false);
    }
  }

  // ---- normalize and store merged-head ctx as bf16 [B*S, DM] ----
#pragma unroll
  for (int r = 0; r < 8; ++r) {
    const float inv = 1.0f / l[r];
    const int srow = qbase + r + 8 * half;
#pragma unroll
    for (int n = 0; n < 4; ++n) {
      const int col = h * DH + n * 16 + lr;
      ctx[(long)(b * SQ + srow) * DM + col] = (__bf16)(acc[n][r] * inv);
    }
  }
}

// =====================================================================
// Kernel 3: output projection  out = ctx @ Wo^T + bo   (bf16 in, f32 out)
// =====================================================================
__global__ void outproj_wmma(const bf16* __restrict__ Xb,  // [MTOT, DM] bf16
                             const float* __restrict__ W,  // [DM, DM]
                             const float* __restrict__ bias,
                             float* __restrict__ out) {    // [MTOT, DM] f32
  const int lane = threadIdx.x & 31;
  const int wave = threadIdx.x >> 5;
  const int half = lane >> 4;
  const int lr   = lane & 15;

  const int tile = blockIdx.x * 8 + wave;
  const int tn   = tile & ((DM / 16) - 1);
  const int tm   = tile >> 6;
  const int m0 = tm * 16, n0 = tn * 16;

  const bf16*  arow = Xb + (long)(m0 + lr) * DM;
  const float* brow = W  + (long)(n0 + lr) * DM;

  v8f c;
#pragma unroll
  for (int r = 0; r < 8; ++r) c[r] = 0.0f;

  for (int kk = 0; kk < DM; kk += 32) {
    const int ka = kk + 16 * half;
    v16bf a = cat8(*(const v8bf*)(arow + ka),
                   *(const v8bf*)(arow + ka + 8));
    v16bf b = cvt16(*(const v16f*)(brow + kk + 16 * half));
    c = __builtin_amdgcn_wmma_f32_16x16x32_bf16(false, a, false, b,
                                                (short)0, c, false, false);
  }

  const float bv = bias[n0 + lr];
#pragma unroll
  for (int r = 0; r < 8; ++r)
    out[(long)(m0 + r + 8 * half) * DM + n0 + lr] = c[r] + bv;
}

// =====================================================================
// launcher
// =====================================================================
extern "C" void kernel_launch(void* const* d_in, const int* in_sizes, int n_in,
                              void* d_out, int out_size, void* d_ws, size_t ws_size,
                              hipStream_t stream) {
  const float* queries = (const float*)d_in[0];
  const float* keys    = (const float*)d_in[1];
  const float* values  = (const float*)d_in[2];
  const float* mask    = (const float*)d_in[3];
  const float* Wq      = (const float*)d_in[4];
  const float* bq      = (const float*)d_in[5];
  const float* Wk      = (const float*)d_in[6];
  const float* bk      = (const float*)d_in[7];
  const float* Wo      = (const float*)d_in[8];
  const float* bo      = (const float*)d_in[9];
  float* out = (float*)d_out;

  // workspace: Q | K | Vt | ctx, each 4M bf16 elements (8 MB) = 32 MB total
  const size_t proj_elems = (size_t)BQ * NH * SQ * DH;   // 4,194,304
  bf16* qws = (bf16*)d_ws;
  bf16* kws = qws + proj_elems;
  bf16* vws = kws + proj_elems;
  bf16* cws = vws + proj_elems;

  const dim3 blk(256);  // 8 waves
  const int projBlocks = (MTOT / 16) * (DM / 16) / 8;    // 2048

  // projections (note: V uses Wk/bk — faithful to the reference's bug);
  // Q is pre-scaled by 1/SCALING = 0.125
  proj_wmma<<<projBlocks, blk, 0, stream>>>(queries, Wq, bq, qws, 0.125f, 0);
  proj_wmma<<<projBlocks, blk, 0, stream>>>(keys,    Wk, bk, kws, 1.0f,   0);
  proj_wmma<<<projBlocks, blk, 0, stream>>>(values,  Wk, bk, vws, 1.0f,   1);

  // streaming attention: B*H * (S/128) blocks, 8 waves x 16 q-rows each
  flash_wmma<<<BQ * NH * (SQ / 128), blk, 0, stream>>>(qws, kws, vws, mask, cws);

  // output projection to f32
  outproj_wmma<<<projBlocks, blk, 0, stream>>>(cws, Wo, bo, out);
}
